// AttentionBlock_51084341019080
// MI455X (gfx1250) — matrix operand order repaired and checked by reading, not compiled, not run on hardware
//
#include <hip/hip_runtime.h>

// ---------------------------------------------------------------------------
// AttentionBlock for MI455X (gfx1250): GroupNorm -> QKV -> flash-attn -> proj
// GEMMs via v_wmma_f32_16x16x32_f16 (f16 in, f32 accumulate), wave32.
// Attention K tiles staged to LDS with global_load_async_to_lds_b128
// (double-buffered, s_wait_asynccnt synchronized).
// ---------------------------------------------------------------------------

typedef _Float16 v16h __attribute__((ext_vector_type(16)));
typedef float    v8f  __attribute__((ext_vector_type(8)));

union FragH { v16h v; float4 f4[2]; _Float16 h[16]; };
union Pack8 { float4 f4; _Float16 h[8]; };

__device__ __forceinline__ v8f wmma32(const v16h& a, const v16h& b, const v8f& c) {
  return __builtin_amdgcn_wmma_f32_16x16x32_f16(false, a, false, b, (short)0, c,
                                                false, false);
}

// Async DMA: global -> LDS, 16 bytes per lane, tracked by ASYNCcnt.
__device__ __forceinline__ void async_ld_b128(unsigned lds_off,
                                              const _Float16* g) {
  asm volatile("global_load_async_to_lds_b128 %0, %1, off"
               :: "v"(lds_off), "v"((unsigned long long)(uintptr_t)g)
               : "memory");
}
#define S_WAIT_ASYNC(n) asm volatile("s_wait_asynccnt %0" :: "i"(n) : "memory")

#define CC   512
#define NH   4
#define CH   128   // head dim
#define NPIX 4096  // H*W
#define QKVROW 1536

// ---------------------------------------------------------------------------
__global__ void f32_to_f16_kernel(const float* __restrict__ s,
                                  _Float16* __restrict__ d, int n) {
  int i = blockIdx.x * blockDim.x + threadIdx.x;
  if (i < n) d[i] = (_Float16)s[i];
}

// ---------------------------------------------------------------------------
// GroupNorm: one block per (b, g). Group = 16 channels x 4096 pixels.
// Writes normalized activations transposed: xnT[b][n][c] (c contiguous, f16).
__global__ void gn_kernel(const float* __restrict__ x,
                          const float* __restrict__ nw,
                          const float* __restrict__ nb,
                          _Float16* __restrict__ xnT) {
  const int blk = blockIdx.x;            // 0..63
  const int b = blk >> 5, g = blk & 31;
  const int tid = threadIdx.x;           // 256 threads
  const float* xg = x + ((size_t)(b * CC + g * 16)) * NPIX;

  float s1 = 0.f, s2 = 0.f;
  for (int i = tid; i < 65536; i += 256) {
    float v = xg[i];
    s1 += v; s2 += v * v;
  }
  __shared__ float r1[256], r2[256];
  r1[tid] = s1; r2[tid] = s2;
  __syncthreads();
  for (int off = 128; off > 0; off >>= 1) {
    if (tid < off) { r1[tid] += r1[tid + off]; r2[tid] += r2[tid + off]; }
    __syncthreads();
  }
  const float inv = 1.0f / 65536.0f;
  const float mean = r1[0] * inv;
  const float var  = r2[0] * inv - mean * mean;
  const float rsig = rsqrtf(var + 1e-5f);

  for (int i = tid; i < 65536; i += 256) {
    int cg = i >> 12, n = i & (NPIX - 1);
    int c = g * 16 + cg;
    float v = (xg[i] - mean) * rsig * nw[c] + nb[c];
    xnT[((size_t)b * NPIX + n) * CC + c] = (_Float16)v;
  }
}

// ---------------------------------------------------------------------------
// QKV GEMM: qkvT[b][n][o] = sum_c W[o][c] * xnT[b][n][c], o in [0,1536).
// q/k rows ((o%384)<256) get the 128^-0.25 attention scale folded in.
__global__ void qkv_gemm(const _Float16* __restrict__ xnT,
                         const _Float16* __restrict__ W,
                         _Float16* __restrict__ qkvT) {
  const int tid = threadIdx.x, lane = tid & 31, wave = tid >> 5;
  const int nb = blockIdx.x * 64;
  const int ob = blockIdx.y * 64 + wave * 16;
  const int b  = blockIdx.z;
  const int hi = (lane < 16) ? 0 : 1;   // lane half
  const int l15 = lane & 15;

  v8f z = {};
  v8f acc[4]; acc[0] = z; acc[1] = z; acc[2] = z; acc[3] = z;

  const _Float16* wrow = W + (size_t)(ob + l15) * CC;
  const _Float16* xb = xnT + (size_t)b * NPIX * CC;

  #pragma unroll 4
  for (int kk = 0; kk < 16; ++kk) {
    FragH a;
    const int aoff = kk * 32 + hi * 8;
    a.f4[0] = *(const float4*)(wrow + aoff);
    a.f4[1] = *(const float4*)(wrow + aoff + 16);
    #pragma unroll
    for (int nt = 0; nt < 4; ++nt) {
      const int n = nb + nt * 16 + l15;
      const _Float16* bp = xb + (size_t)n * CC + kk * 32 + hi * 8;
      FragH bf;
      bf.f4[0] = *(const float4*)(bp);
      bf.f4[1] = *(const float4*)(bp + 16);
      acc[nt] = wmma32(a.v, bf.v, acc[nt]);
    }
  }

  const float SCALE = 0.29730177875068026f;  // 128^-0.25
  const int o0 = ob + hi * 8;
  const float s = ((o0 % 384) < 256) ? SCALE : 1.0f;
  #pragma unroll
  for (int nt = 0; nt < 4; ++nt) {
    const int n = nb + nt * 16 + l15;
    Pack8 p;
    #pragma unroll
    for (int r = 0; r < 8; ++r) p.h[r] = (_Float16)(acc[nt][r] * s);
    *(float4*)(qkvT + ((size_t)b * NPIX + n) * QKVROW + o0) = p.f4;
  }
}

// ---------------------------------------------------------------------------
// Flash attention. Block = 128 threads = 4 waves; grid = (N/64, B*NH).
// Tr=64 query rows per block (16 per wave), Tc=32 keys per iteration.
// Q frags resident in VGPRs; K tiles async-DMA'd to LDS (double buffered);
// V staged via registers with transpose.
#define TC   32
#define NBLK (NPIX / TC)
#define SST  33   // lds_S row stride (f32)  -> conflict-free column access
#define PST  40   // lds_P row stride (f16)  -> 16B-aligned, conflict-spread
#define VST  40   // lds_V row stride (f16)
#define KST  136  // lds_K row stride (f16): 272B rows, 16B-aligned, 4-bank skew

__global__ void attn_kernel(const _Float16* __restrict__ qkvT,
                            _Float16* __restrict__ hT) {
  const int tid = threadIdx.x, lane = tid & 31, wave = tid >> 5;
  const int tb = blockIdx.x * 64;
  const int head = blockIdx.y;               // 0..7
  const int b = head >> 2, h = head & 3;
  const int hi = (lane < 16) ? 0 : 1;
  const int l15 = lane & 15;

  __shared__ __align__(16) float    lds_S[64 * SST];
  __shared__ __align__(16) _Float16 lds_P[64 * PST];
  __shared__ __align__(16) _Float16 lds_V[CH * VST];
  __shared__ __align__(16) _Float16 lds_K[2][TC][KST];
  __shared__ float lds_alpha[64];
  __shared__ float lds_linv[64];

  // head h occupies rows [384h, 384h+128)=q, +128=k, +256=v of qkvT
  const _Float16* qbase = qkvT + ((size_t)b * NPIX) * QKVROW + h * 384;

  // per-thread async-copy slice of a K tile: 4 x b128 (32 rows of 256B each
  // handled by 16 lanes); natural [s][c] layout == WMMA B-fragment layout.
  const unsigned ldsK0 = (unsigned)(uintptr_t)(&lds_K[0][0][0]);
  const int krow = tid >> 4;          // 0..7   (base row, step 8)
  const int kcol = (tid & 15) * 8;    // halves within row
  auto issueK = [&](int sblk, int buf) {
    const _Float16* kb = qbase + 128;
    #pragma unroll
    for (int j = 0; j < 4; ++j) {
      const int row = krow + j * 8;
      const _Float16* g = kb + (size_t)(sblk * TC + row) * QKVROW + kcol;
      const unsigned lo =
          ldsK0 + ((unsigned)buf * TC * KST + (unsigned)row * KST +
                   (unsigned)kcol) * 2u;
      async_ld_b128(lo, g);
    }
  };

  // resident Q fragments (A layout: M=t, K=c; 4 k-steps of 32)
  FragH qf[4];
  {
    const int t = tb + wave * 16 + l15;
    const _Float16* qrow = qbase + (size_t)t * QKVROW;
    #pragma unroll
    for (int kk = 0; kk < 4; ++kk) {
      const int o = kk * 32 + hi * 8;
      qf[kk].f4[0] = *(const float4*)(qrow + o);
      qf[kk].f4[1] = *(const float4*)(qrow + o + 16);
    }
  }

  v8f z = {};
  v8f oacc[8];
  #pragma unroll
  for (int i = 0; i < 8; ++i) oacc[i] = z;

  float m = -1e30f, l = 0.0f;  // per-row softmax state (threads 0..63)

  issueK(0, 0);  // prologue: first K tile in flight

  for (int sb = 0; sb < NBLK; ++sb) {
    const int cur = sb & 1;
    const int s0 = sb * TC;
    __syncthreads();  // WAR: lds_P / lds_V / lds_K[cur^1] free for rewrite

    // prefetch next K tile into the other buffer (wraps on last iteration
    // so the <=4 async wait below always has 8 outstanding to cut through)
    issueK((sb + 1) & (NBLK - 1), cur ^ 1);

    // stage V block into LDS transposed: lds_V[c][s]
    {
      const int sl = tid & 31;
      const int cb = (tid >> 5) * 32;
      const _Float16* vrow = qbase + (size_t)(s0 + sl) * QKVROW + 256;
      #pragma unroll
      for (int j = 0; j < 4; ++j) {
        Pack8 u; u.f4 = *(const float4*)(vrow + cb + 8 * j);
        #pragma unroll
        for (int e = 0; e < 8; ++e)
          lds_V[(cb + 8 * j + e) * VST + sl] = u.h[e];
      }
    }

    // current tile's 4 asyncs (issued last iteration / prologue) complete;
    // the 4 just-issued may stay in flight (in-order per wave).
    S_WAIT_ASYNC(4);
    __syncthreads();  // publish all waves' K copies

    // S = (scaled Q)^T (scaled K): 2 s-tiles x 4 k-steps, K from LDS
    v8f sacc[2]; sacc[0] = z; sacc[1] = z;
    #pragma unroll
    for (int kk = 0; kk < 4; ++kk) {
      #pragma unroll
      for (int st = 0; st < 2; ++st) {
        const _Float16* kp = &lds_K[cur][st * 16 + l15][kk * 32 + hi * 8];
        FragH bf;
        bf.f4[0] = *(const float4*)(kp);
        bf.f4[1] = *(const float4*)(kp + 16);
        sacc[st] = wmma32(qf[kk].v, bf.v, sacc[st]);
      }
    }
    #pragma unroll
    for (int st = 0; st < 2; ++st)
      #pragma unroll
      for (int r = 0; r < 8; ++r)
        lds_S[(wave * 16 + hi * 8 + r) * SST + st * 16 + l15] = sacc[st][r];

    __syncthreads();

    // online softmax: one thread per query row
    if (tid < 64) {
      const float* srow = &lds_S[tid * SST];
      float mn = m;
      #pragma unroll 8
      for (int s = 0; s < TC; ++s) mn = fmaxf(mn, srow[s]);
      const float alpha = __expf(m - mn);
      float ssum = 0.f;
      _Float16* prow = &lds_P[tid * PST];
      #pragma unroll 8
      for (int s = 0; s < TC; ++s) {
        const float p = __expf(srow[s] - mn);
        ssum += p;
        prow[s] = (_Float16)p;
      }
      lds_alpha[tid] = alpha;
      m = mn;
      l = l * alpha + ssum;
    }
    __syncthreads();

    // O = alpha*O + V * P   (D layout: M=c, N=t; single K=32 step)
    const float alpha = lds_alpha[wave * 16 + l15];
    FragH pf;
    {
      const _Float16* pp = &lds_P[(wave * 16 + l15) * PST + hi * 8];
      pf.f4[0] = *(const float4*)(pp);
      pf.f4[1] = *(const float4*)(pp + 16);
    }
    #pragma unroll
    for (int ct = 0; ct < 8; ++ct) {
      oacc[ct] = oacc[ct] * alpha;
      FragH vf;
      const _Float16* vp = &lds_V[(ct * 16 + l15) * VST + hi * 8];
      vf.f4[0] = *(const float4*)(vp);
      vf.f4[1] = *(const float4*)(vp + 16);
      oacc[ct] = wmma32(vf.v, pf.v, oacc[ct]);
    }
  }

  S_WAIT_ASYNC(0);  // drain wrapped prefetch before kernel exit
  __syncthreads();
  if (tid < 64) lds_linv[tid] = 1.0f / l;
  __syncthreads();

  // write hT[b][t][h*128 + c] (f16, c contiguous)
  {
    const float rl = lds_linv[wave * 16 + l15];
    const int t = tb + wave * 16 + l15;
    _Float16* orow = hT + ((size_t)b * NPIX + t) * CC + h * CH;
    #pragma unroll
    for (int ct = 0; ct < 8; ++ct) {
      Pack8 p;
      #pragma unroll
      for (int r = 0; r < 8; ++r) p.h[r] = (_Float16)(oacc[ct][r] * rl);
      *(float4*)(orow + ct * 16 + hi * 8) = p.f4;
    }
  }
}

// ---------------------------------------------------------------------------
// Output projection + bias + residual:
// out[b][o][n] = sum_c Wp[o][c]*hT[b][n][c] + bias[o] + x[b][o][n]
__global__ void proj_gemm(const _Float16* __restrict__ hT,
                          const _Float16* __restrict__ W,
                          const float* __restrict__ bias,
                          const float* __restrict__ x,
                          float* __restrict__ out) {
  const int tid = threadIdx.x, lane = tid & 31, wave = tid >> 5;
  const int nb = blockIdx.x * 64;
  const int ob = blockIdx.y * 64 + wave * 16;
  const int b  = blockIdx.z;
  const int hi = (lane < 16) ? 0 : 1;
  const int l15 = lane & 15;

  v8f z = {};
  v8f acc[4]; acc[0] = z; acc[1] = z; acc[2] = z; acc[3] = z;

  const _Float16* wrow = W + (size_t)(ob + l15) * CC;
  const _Float16* hb = hT + (size_t)b * NPIX * CC;

  #pragma unroll 4
  for (int kk = 0; kk < 16; ++kk) {
    FragH a;
    const int aoff = kk * 32 + hi * 8;
    a.f4[0] = *(const float4*)(wrow + aoff);
    a.f4[1] = *(const float4*)(wrow + aoff + 16);
    #pragma unroll
    for (int nt = 0; nt < 4; ++nt) {
      const int n = nb + nt * 16 + l15;
      const _Float16* bp = hb + (size_t)n * CC + kk * 32 + hi * 8;
      FragH bf;
      bf.f4[0] = *(const float4*)(bp);
      bf.f4[1] = *(const float4*)(bp + 16);
      acc[nt] = wmma32(a.v, bf.v, acc[nt]);
    }
  }

  const int o0 = ob + hi * 8;
  #pragma unroll
  for (int nt = 0; nt < 4; ++nt) {
    const int n = nb + nt * 16 + l15;
    #pragma unroll
    for (int r = 0; r < 8; ++r) {
      const int o = o0 + r;
      const size_t idx = ((size_t)(b * CC + o)) * NPIX + n;
      out[idx] = acc[nt][r] + bias[o] + x[idx];
    }
  }
}

// ---------------------------------------------------------------------------
extern "C" void kernel_launch(void* const* d_in, const int* in_sizes, int n_in,
                              void* d_out, int out_size, void* d_ws, size_t ws_size,
                              hipStream_t stream) {
  (void)in_sizes; (void)n_in; (void)out_size; (void)ws_size;

  const float* x      = (const float*)d_in[0];
  const float* norm_w = (const float*)d_in[1];
  const float* norm_b = (const float*)d_in[2];
  const float* w_qkv  = (const float*)d_in[3];
  const float* w_proj = (const float*)d_in[4];
  const float* b_proj = (const float*)d_in[5];
  float* out = (float*)d_out;

  // workspace carve (bytes), all offsets 256B-aligned
  char* ws = (char*)d_ws;
  _Float16* xnT   = (_Float16*)(ws);                     //  8,388,608
  _Float16* hT    = (_Float16*)(ws + 8388608u);          //  8,388,608
  _Float16* qkvT  = (_Float16*)(ws + 16777216u);         // 25,165,824
  _Float16* wq16  = (_Float16*)(ws + 41943040u);         //  1,572,864
  _Float16* wp16  = (_Float16*)(ws + 43515904u);         //    524,288
  // total: 44,040,192 bytes

  f32_to_f16_kernel<<<dim3(3072), 256, 0, stream>>>(w_qkv, wq16, 3 * CC * CC);
  f32_to_f16_kernel<<<dim3(1024), 256, 0, stream>>>(w_proj, wp16, CC * CC);

  gn_kernel<<<dim3(64), 256, 0, stream>>>(x, norm_w, norm_b, xnT);

  qkv_gemm<<<dim3(NPIX / 64, (3 * CC) / 64, 2), 128, 0, stream>>>(xnT, wq16, qkvT);

  attn_kernel<<<dim3(NPIX / 64, 2 * NH), 128, 0, stream>>>(qkvT, hT);

  proj_gemm<<<dim3(NPIX / 64, CC / 64, 2), 128, 0, stream>>>(hT, wp16, b_proj, x, out);
}
